// MatAnyone_60292750901478
// MI455X (gfx1250) — compile-verified, matrix operand order checked by reading, not run on hardware
//
#include <hip/hip_runtime.h>
#include <hip/hip_bf16.h>

// MatAnyone memory-readout fused as flash-attention on gfx1250 WMMA (bf16).
// B=2, CK=64, CV=256, T=8, H=W=48 -> HW=2304, THW=18432.
// One WG = 32 query columns, 8 waves; loop tiles 128 keys; cv split 32/wave.

typedef __attribute__((ext_vector_type(16))) __bf16 v16bf;
typedef __attribute__((ext_vector_type(8)))  float  v8f;

constexpr int kB    = 2;
constexpr int kCK   = 64;
constexpr int kCV   = 256;
constexpr int kHW   = 48 * 48;     // 2304
constexpr int kTHW  = 8 * kHW;     // 18432
constexpr int kKT   = 128;         // keys per mainloop iteration
constexpr int kNQ   = 32;          // query columns per workgroup
constexpr int kSROW = 132;         // padded LDS stride (f32 score tile)
constexpr int kPROW = 136;         // padded LDS stride (bf16 p tile)

union V16U { v16bf v; float4 f[2]; };
union P8U  { __bf16 h[8]; float4 f; };
union P4U  { __bf16 h[4]; float2 f; };

// ---------------------------------------------------------------------------
// Prep 1: mkh[b][t][kk] (bf16), kk<64 -> mk[c=kk][t], kk>=64 -> mk[c]^2.
// ---------------------------------------------------------------------------
__global__ __launch_bounds__(256, 1)
void pack_mk(const float* __restrict__ mk, __bf16* __restrict__ mkh) {
  const int b  = blockIdx.y;
  const int t0 = blockIdx.x * 64;
  __shared__ float tile[64][65];

  const int ti = threadIdx.x & 63;
  for (int c = threadIdx.x >> 6; c < 64; c += 4)
    tile[c][ti] = mk[((long)b * kCK + c) * kTHW + t0 + ti];
  __syncthreads();

  const int i = threadIdx.x >> 2;          // t row 0..63
  const int q = threadIdx.x & 3;           // kk quarter
  alignas(16) __bf16 outv[32];
#pragma unroll
  for (int x = 0; x < 32; ++x) {
    int kk  = q * 32 + x;
    float v = tile[kk & 63][i];
    if (kk >= 64) v *= v;
    outv[x] = (__bf16)v;
  }
  __bf16* dst = mkh + ((long)(b * kTHW + t0 + i) * 128 + q * 32);
#pragma unroll
  for (int x = 0; x < 4; ++x)
    *(float4*)(dst + x * 8) = *(float4*)(outv + x * 8);
}

// ---------------------------------------------------------------------------
// Prep 2: mvh = bf16(msk_value), same (b,cv,t) layout. n divisible by 4.
// ---------------------------------------------------------------------------
__global__ __launch_bounds__(256, 1)
void pack_mv(const float* __restrict__ mv, __bf16* __restrict__ mvh, int n) {
  int i = (blockIdx.x * 256 + threadIdx.x) * 4;
  if (i < n) {
    float4 v = *(const float4*)(mv + i);
    P4U o;
    o.h[0] = (__bf16)v.x; o.h[1] = (__bf16)v.y;
    o.h[2] = (__bf16)v.z; o.h[3] = (__bf16)v.w;
    *(float2*)(mvh + i) = o.f;
  }
}

// ---------------------------------------------------------------------------
// Main fused kernel. PACKED: A-operands come pre-packed bf16 from d_ws.
// ---------------------------------------------------------------------------
template <bool PACKED>
__global__ __launch_bounds__(256, 1)
void matanyone_flash_wmma(const float* __restrict__ qk,   // (B,CK,H,W)
                          const float* __restrict__ qe,   // (B,CK,H,W)
                          const float* __restrict__ mk,   // (B,CK,T,H,W)
                          const float* __restrict__ ms,   // (B,1,T,H,W)
                          const float* __restrict__ mv,   // (B,1,CV,T,H,W)
                          const float* __restrict__ up,   // (B,1,H,W)
                          const __bf16* __restrict__ mkh, // packed [mk;mk^2]
                          const __bf16* __restrict__ mvh, // packed mv
                          float*        __restrict__ out) // (B,1,CV,H,W)
{
  const int b    = blockIdx.y;
  const int n0   = blockIdx.x * kNQ;
  const int tid  = threadIdx.x;
  const int lane = tid & 31;
  const int wv   = tid >> 5;
  const int l16  = lane & 15;
  const int half = lane >> 4;

  __shared__ float  S[kNQ * kSROW];   // f32 scores: 32 cols x 128 rows
  __shared__ __bf16 Pb[kNQ * kPROW];  // bf16 softmax numerators
  __shared__ float  PM[256], PS[256];
  __shared__ float  Mrun[kNQ], Lrun[kNQ], CORR[kNQ], MNEW[kNQ], BSQ[kNQ];

  if (tid < kNQ) {
    const int n = n0 + tid;
    float s = 0.f;
    for (int c = 0; c < kCK; ++c) {
      float k = qk[(b * kCK + c) * kHW + n];
      float e = qe[(b * kCK + c) * kHW + n];
      s += e * k * k;
    }
    BSQ[tid]  = s;
    Mrun[tid] = -3.0e38f;
    Lrun[tid] = 0.f;
  }

  // Resident B fragments for the score GEMM: B = [2*qk*qe ; -qe], 128(K)x16(N)
  v16bf bq[2][4];
#pragma unroll
  for (int nt = 0; nt < 2; ++nt) {
    const int n = n0 + nt * 16 + l16;
#pragma unroll
    for (int j = 0; j < 4; ++j) {
#pragma unroll
      for (int e = 0; e < 16; ++e) {
        int kk = j * 32 + ((e < 8) ? (half * 8 + e) : (16 + half * 8 + (e - 8)));
        float v;
        if (kk < 64) {
          v = 2.0f * qk[(b * kCK + kk) * kHW + n] * qe[(b * kCK + kk) * kHW + n];
        } else {
          v = -qe[(b * kCK + (kk - 64)) * kHW + n];
        }
        bq[nt][j][e] = (__bf16)v;
      }
    }
  }

  __syncthreads();
  const float bsq0 = BSQ[l16];
  const float bsq1 = BSQ[16 + l16];

  v8f o00 = {0,0,0,0,0,0,0,0};   // cv [wv*32   ,+16) x cols [n0   ,+16)
  v8f o01 = {0,0,0,0,0,0,0,0};   // cv [wv*32   ,+16) x cols [n0+16,+16)
  v8f o10 = {0,0,0,0,0,0,0,0};   // cv [wv*32+16,+16) x cols [n0   ,+16)
  v8f o11 = {0,0,0,0,0,0,0,0};   // cv [wv*32+16,+16) x cols [n0+16,+16)

  const float*  mkB  = mk  + (long)b * kCK * kTHW;
  const float*  msB  = ms  + (long)b * kTHW;
  const float*  mvB  = mv  + (long)b * kCV * kTHW;
  const __bf16* mvhB = PACKED ? (mvh + (long)b * kCV * kTHW) : nullptr;

  for (int kt = 0; kt < kTHW; kt += kKT) {
    __syncthreads();                       // previous readout done with S/Pb

    // ---- scores: wave wv covers keys [kt + wv*16, +16), K = 128 ----
    const int t0 = kt + wv * 16;
    v8f sc0 = {0,0,0,0,0,0,0,0};
    v8f sc1 = {0,0,0,0,0,0,0,0};
    if (PACKED) {
      const __bf16* arow = mkh + (long)(b * kTHW + t0 + l16) * 128 + half * 8;
#pragma unroll
      for (int j = 0; j < 4; ++j) {
        V16U a;
        a.f[0] = *(const float4*)(arow + j * 32);
        a.f[1] = *(const float4*)(arow + j * 32 + 16);
        sc0 = __builtin_amdgcn_wmma_f32_16x16x32_bf16(
                  false, a.v, false, bq[0][j], (short)0, sc0, false, false);
        sc1 = __builtin_amdgcn_wmma_f32_16x16x32_bf16(
                  false, a.v, false, bq[1][j], (short)0, sc1, false, false);
      }
    } else {
      const int t = t0 + l16;
#pragma unroll
      for (int j = 0; j < 4; ++j) {
        v16bf a;
#pragma unroll
        for (int e = 0; e < 16; ++e) {
          int kk = j * 32 + ((e < 8) ? (half * 8 + e) : (16 + half * 8 + (e - 8)));
          float x = mkB[(kk & 63) * kTHW + t];
          if (kk >= 64) x *= x;
          a[e] = (__bf16)x;
        }
        sc0 = __builtin_amdgcn_wmma_f32_16x16x32_bf16(
                  false, a, false, bq[0][j], (short)0, sc0, false, false);
        sc1 = __builtin_amdgcn_wmma_f32_16x16x32_bf16(
                  false, a, false, bq[1][j], (short)0, sc1, false, false);
      }
    }

    // similarity = (score - b_sq[n]) * ms[t] / 8 ; write tiles to LDS (f32)
    {
      const float4 m0 = *(const float4*)(msB + t0 + half * 8);
      const float4 m1 = *(const float4*)(msB + t0 + half * 8 + 4);
      const float msv[8] = {m0.x, m0.y, m0.z, m0.w, m1.x, m1.y, m1.z, m1.w};
      float s0[8], s1[8];
#pragma unroll
      for (int r = 0; r < 8; ++r) {
        s0[r] = (sc0[r] - bsq0) * msv[r] * 0.125f;
        s1[r] = (sc1[r] - bsq1) * msv[r] * 0.125f;
      }
      float* d0 = &S[l16 * kSROW + wv * 16 + half * 8];
      float* d1 = &S[(16 + l16) * kSROW + wv * 16 + half * 8];
      *(float4*)(d0 + 0) = make_float4(s0[0], s0[1], s0[2], s0[3]);
      *(float4*)(d0 + 4) = make_float4(s0[4], s0[5], s0[6], s0[7]);
      *(float4*)(d1 + 0) = make_float4(s1[0], s1[1], s1[2], s1[3]);
      *(float4*)(d1 + 4) = make_float4(s1[4], s1[5], s1[6], s1[7]);
    }
    __syncthreads();

    const int col = tid & 31;              // 32 query columns
    const int seg = tid >> 5;              // 8 row segments of 16 rows

    {
      float pm = -3.0e38f;
#pragma unroll
      for (int r = 0; r < 16; ++r)
        pm = fmaxf(pm, S[col * kSROW + seg * 16 + r]);
      PM[seg * 32 + col] = pm;
    }
    __syncthreads();
    if (tid < kNQ) {
      float mt = -3.0e38f;
      for (int sg = 0; sg < 8; ++sg) mt = fmaxf(mt, PM[sg * 32 + tid]);
      float Mo = Mrun[tid];
      float Mn = fmaxf(Mo, mt);
      CORR[tid] = __expf(Mo - Mn);
      MNEW[tid] = Mn;
      Mrun[tid] = Mn;
    }
    __syncthreads();
    // exponentiate; convert to bf16 once and stage into Pb for B-fragments
    {
      const float mn = MNEW[col];
      float ps = 0.f;
      P8U pw0, pw1;
#pragma unroll
      for (int r = 0; r < 8; ++r) {
        float v = __expf(S[col * kSROW + seg * 16 + r] - mn);
        pw0.h[r] = (__bf16)v;
        ps += v;
      }
#pragma unroll
      for (int r = 0; r < 8; ++r) {
        float v = __expf(S[col * kSROW + seg * 16 + 8 + r] - mn);
        pw1.h[r] = (__bf16)v;
        ps += v;
      }
      *(float4*)&Pb[col * kPROW + seg * 16]     = pw0.f;
      *(float4*)&Pb[col * kPROW + seg * 16 + 8] = pw1.f;
      PS[seg * 32 + col] = ps;
    }
    __syncthreads();                       // Pb final
    if (tid < kNQ) {
      float s = 0.f;
      for (int sg = 0; sg < 8; ++sg) s += PS[sg * 32 + tid];
      Lrun[tid] = Lrun[tid] * CORR[tid] + s;
    }

    // ---- readout: O = O*corr + mv_tile(16x128) x P(128x16) ----
    const float cr0 = CORR[l16];
    const float cr1 = CORR[16 + l16];
#pragma unroll
    for (int r = 0; r < 8; ++r) {
      o00[r] *= cr0; o10[r] *= cr0;
      o01[r] *= cr1; o11[r] *= cr1;
    }

    V16U bp[2][4];
#pragma unroll
    for (int nt = 0; nt < 2; ++nt) {
#pragma unroll
      for (int j = 0; j < 4; ++j) {
        const __bf16* pb = &Pb[(nt * 16 + l16) * kPROW + j * 32 + half * 8];
        bp[nt][j].f[0] = *(const float4*)(pb);
        bp[nt][j].f[1] = *(const float4*)(pb + 16);
      }
    }
#pragma unroll
    for (int ct = 0; ct < 2; ++ct) {
      const int cvb = wv * 32 + ct * 16;
      if (PACKED) {
        const __bf16* rp = mvhB + (long)(cvb + l16) * kTHW + kt + half * 8;
        if (kt + kKT < kTHW) __builtin_prefetch(rp + kKT, 0, 1);
#pragma unroll
        for (int j = 0; j < 4; ++j) {
          V16U a;
          a.f[0] = *(const float4*)(rp + j * 32);
          a.f[1] = *(const float4*)(rp + j * 32 + 16);
          v8f* oa = (ct == 0) ? &o00 : &o10;
          v8f* ob = (ct == 0) ? &o01 : &o11;
          *oa = __builtin_amdgcn_wmma_f32_16x16x32_bf16(
                    false, a.v, false, bp[0][j].v, (short)0, *oa, false, false);
          *ob = __builtin_amdgcn_wmma_f32_16x16x32_bf16(
                    false, a.v, false, bp[1][j].v, (short)0, *ob, false, false);
        }
      } else {
        const float* rp = mvB + (long)(cvb + l16) * kTHW + kt + half * 8;
#pragma unroll
        for (int j = 0; j < 4; ++j) {
          float4 a0 = *(const float4*)(rp + j * 32);
          float4 a1 = *(const float4*)(rp + j * 32 + 4);
          float4 a2 = *(const float4*)(rp + j * 32 + 16);
          float4 a3 = *(const float4*)(rp + j * 32 + 20);
          v16bf a;
          a[0]=(__bf16)a0.x; a[1]=(__bf16)a0.y; a[2]=(__bf16)a0.z; a[3]=(__bf16)a0.w;
          a[4]=(__bf16)a1.x; a[5]=(__bf16)a1.y; a[6]=(__bf16)a1.z; a[7]=(__bf16)a1.w;
          a[8]=(__bf16)a2.x; a[9]=(__bf16)a2.y; a[10]=(__bf16)a2.z; a[11]=(__bf16)a2.w;
          a[12]=(__bf16)a3.x; a[13]=(__bf16)a3.y; a[14]=(__bf16)a3.z; a[15]=(__bf16)a3.w;
          v8f* oa = (ct == 0) ? &o00 : &o10;
          v8f* ob = (ct == 0) ? &o01 : &o11;
          *oa = __builtin_amdgcn_wmma_f32_16x16x32_bf16(
                    false, a, false, bp[0][j].v, (short)0, *oa, false, false);
          *ob = __builtin_amdgcn_wmma_f32_16x16x32_bf16(
                    false, a, false, bp[1][j].v, (short)0, *ob, false, false);
        }
      }
    }
  }

  __syncthreads();
#pragma unroll
  for (int nt = 0; nt < 2; ++nt) {
    const float linv = 1.0f / Lrun[nt * 16 + l16];
    const int   n    = n0 + nt * 16 + l16;
    const float pu   = up[b * kHW + n];
#pragma unroll
    for (int ct = 0; ct < 2; ++ct) {
#pragma unroll
      for (int r = 0; r < 8; ++r) {
        const int cv = wv * 32 + ct * 16 + half * 8 + r;   // D row = r + half*8
        float o;
        if (nt == 0) o = (ct == 0 ? o00[r] : o10[r]) * linv;
        else         o = (ct == 0 ? o01[r] : o11[r]) * linv;
        float lastv = mvB[(long)cv * kTHW + 7 * kHW + n];  // msk_value[...,T-1]
        out[(b * kCV + cv) * kHW + n] = o * pu + lastv * (1.0f - pu);
      }
    }
  }
}

extern "C" void kernel_launch(void* const* d_in, const int* in_sizes, int n_in,
                              void* d_out, int out_size, void* d_ws, size_t ws_size,
                              hipStream_t stream) {
  const float* qk = (const float*)d_in[0];   // query_key
  const float* qe = (const float*)d_in[1];   // query_selection
  const float* mk = (const float*)d_in[2];   // memory_key
  const float* ms = (const float*)d_in[3];   // memory_shrinkage
  const float* mv = (const float*)d_in[4];   // msk_value
  const float* up = (const float*)d_in[5];   // uncert_prob
  float* out = (float*)d_out;

  const size_t mkh_bytes = (size_t)kB * kTHW * 128 * sizeof(__bf16); // ~9.4 MB
  const size_t mvh_bytes = (size_t)kB * kCV * kTHW * sizeof(__bf16); // ~18.9 MB
  dim3 grid(kHW / kNQ, kB);                  // (72, 2)

  if (ws_size >= mkh_bytes + mvh_bytes) {
    __bf16* mkh = (__bf16*)d_ws;
    __bf16* mvh = (__bf16*)((char*)d_ws + mkh_bytes);
    pack_mk<<<dim3(kTHW / 64, kB), 256, 0, stream>>>(mk, mkh);
    const int nmv = kB * kCV * kTHW;
    pack_mv<<<(nmv / 4 + 255) / 256, 256, 0, stream>>>(mv, mvh, nmv);
    matanyone_flash_wmma<true><<<grid, 256, 0, stream>>>(
        qk, qe, mk, ms, mv, up, mkh, mvh, out);
  } else {
    matanyone_flash_wmma<false><<<grid, 256, 0, stream>>>(
        qk, qe, mk, ms, mv, up, nullptr, nullptr, out);
  }
}
